// GenericCrossFieldAttn_42442866819676
// MI455X (gfx1250) — matrix operand order repaired and checked by reading, not compiled, not run on hardware
//
#include <hip/hip_runtime.h>
#include <hip/hip_bf16.h>
#include <math.h>

typedef __attribute__((ext_vector_type(16))) _Float16 v16h;
typedef __attribute__((ext_vector_type(8)))  float    v8f;

#define E_   512
#define NH_  8
#define HD_  64
#define NT_  32
#define B_   8
#define N_   4096
#define N2_  8192

static __device__ __forceinline__ float sigmoidf_(float x) { return 1.f / (1.f + __expf(-x)); }

// ---- CDNA5 async global->LDS copy (ASYNCcnt path), 16B per lane -----------
static __device__ __forceinline__ void async_copy_b128(void* lds, const void* g) {
  // VDST = VGPR with LDS byte address (low 32 bits of generic addr), VADDR = 64b global
  asm volatile("global_load_async_to_lds_b128 %0, %1, off"
               :: "v"((unsigned)(size_t)lds), "v"(g)
               : "memory");
}
static __device__ __forceinline__ void wait_async0() {
#if __has_builtin(__builtin_amdgcn_s_wait_asynccnt)
  __builtin_amdgcn_s_wait_asynccnt(0);
#else
  asm volatile("s_wait_asynccnt 0" ::: "memory");
#endif
}

// ---------------------------------------------------------------------------
// f32-operand GEMM (small M): C[M,Nn] = act(A @ W^T + bias). WMMA f16, f32 acc.
// ---------------------------------------------------------------------------
template <int ACT>  // 0=none 1=sigmoid 2=silu
__global__ __launch_bounds__(256) void gemm_wmma(const float* __restrict__ A,
                                                 const float* __restrict__ W,
                                                 const float* __restrict__ bias,
                                                 float* __restrict__ C,
                                                 int M, int Nn, int K) {
  __shared__ _Float16 As[128][40];
  __shared__ _Float16 Ws[64][40];
  const int tid  = threadIdx.x;
  const int lane = tid & 31, wave = tid >> 5;
  const int m0 = blockIdx.y * 128;
  const int n0 = blockIdx.x * 64;
  const int mloc = wave * 16 + (lane & 15);
  const int bk0  = (lane < 16) ? 0 : 8;
  const int kk0  = (lane < 16) ? 0 : 16;
  v8f acc[4] = {};

  for (int kt = 0; kt < K; kt += 32) {
    {
      int r = tid >> 1, c = (tid & 1) * 16;
      int gm = m0 + r;
      const float* src = A + (size_t)gm * K + kt + c;
      if (gm < M) {
#pragma unroll
        for (int i = 0; i < 16; ++i) As[r][c + i] = (_Float16)src[i];
      } else {
#pragma unroll
        for (int i = 0; i < 16; ++i) As[r][c + i] = (_Float16)0.f;
      }
    }
    {
      int r = tid >> 2, c = (tid & 3) * 8;
      const float* src = W + (size_t)(n0 + r) * K + kt + c;
#pragma unroll
      for (int i = 0; i < 8; ++i) Ws[r][c + i] = (_Float16)src[i];
    }
    __syncthreads();
    v16h a;
#pragma unroll
    for (int i = 0; i < 4; ++i) { a[2 * i] = As[mloc][bk0 + 2 * i];      a[2 * i + 1] = As[mloc][bk0 + 2 * i + 1]; }
#pragma unroll
    for (int i = 0; i < 4; ++i) { a[8 + 2 * i] = As[mloc][bk0 + 16 + 2 * i]; a[9 + 2 * i] = As[mloc][bk0 + 16 + 2 * i + 1]; }
#pragma unroll
    for (int t = 0; t < 4; ++t) {
      int n = t * 16 + (lane & 15);
      v16h bf;
#pragma unroll
      for (int j = 0; j < 16; ++j) bf[j] = Ws[n][bk0 + j + (j & 8)];
      acc[t] = __builtin_amdgcn_wmma_f32_16x16x32_f16(false, a, false, bf, (short)0, acc[t], false, false);
    }
    __syncthreads();
  }
#pragma unroll
  for (int t = 0; t < 4; ++t)
#pragma unroll
    for (int j = 0; j < 8; ++j) {
      int row = m0 + wave * 16 + ((lane < 16) ? j : j + 8);
      int col = n0 + t * 16 + (lane & 15);
      if (row < M) {
        float v = acc[t][j] + bias[col];
        if (ACT == 1) v = sigmoidf_(v);
        else if (ACT == 2) v = v * sigmoidf_(v);
        C[(size_t)row * Nn + col] = v;
      }
    }
}

// ---------------------------------------------------------------------------
// f16-operand GEMM (big M, M%128==0): async global->LDS tile copies.
// C = act(A @ W^T + bias); OUT16 selects f16 or f32 output.
// ---------------------------------------------------------------------------
template <int ACT, int OUT16>
__global__ __launch_bounds__(256) void gemm_f16a(const _Float16* __restrict__ A,
                                                 const _Float16* __restrict__ W,
                                                 const float* __restrict__ bias,
                                                 void* __restrict__ Cv,
                                                 int M, int Nn, int K) {
  __shared__ _Float16 As[128][48];  // 96B row stride: 16B-aligned async stores
  __shared__ _Float16 Ws[64][48];
  const int tid  = threadIdx.x;
  const int lane = tid & 31, wave = tid >> 5;
  const int m0 = blockIdx.y * 128;
  const int n0 = blockIdx.x * 64;
  const int mloc = wave * 16 + (lane & 15);
  const int bk0  = (lane < 16) ? 0 : 8;
  const int kk0  = (lane < 16) ? 0 : 16;
  v8f acc[4] = {};

  for (int kt = 0; kt < K; kt += 32) {
    // A tile: 128x32 halves = 512 x 16B chunks, 2 per thread
#pragma unroll
    for (int i = 0; i < 2; ++i) {
      int c2 = tid + 256 * i;
      int r = c2 >> 2, cc = (c2 & 3) * 8;
      async_copy_b128(&As[r][cc], A + (size_t)(m0 + r) * K + kt + cc);
    }
    // W tile: 64x32 halves = 256 x 16B chunks, 1 per thread
    {
      int r = tid >> 2, cc = (tid & 3) * 8;
      async_copy_b128(&Ws[r][cc], W + (size_t)(n0 + r) * K + kt + cc);
    }
    wait_async0();
    __syncthreads();

    v16h a;
#pragma unroll
    for (int i = 0; i < 4; ++i) { a[2 * i] = As[mloc][bk0 + 2 * i];      a[2 * i + 1] = As[mloc][bk0 + 2 * i + 1]; }
#pragma unroll
    for (int i = 0; i < 4; ++i) { a[8 + 2 * i] = As[mloc][bk0 + 16 + 2 * i]; a[9 + 2 * i] = As[mloc][bk0 + 16 + 2 * i + 1]; }
#pragma unroll
    for (int t = 0; t < 4; ++t) {
      int n = t * 16 + (lane & 15);
      v16h bf;
#pragma unroll
      for (int j = 0; j < 16; ++j) bf[j] = Ws[n][bk0 + j + (j & 8)];
      acc[t] = __builtin_amdgcn_wmma_f32_16x16x32_f16(false, a, false, bf, (short)0, acc[t], false, false);
    }
    __syncthreads();
  }
#pragma unroll
  for (int t = 0; t < 4; ++t)
#pragma unroll
    for (int j = 0; j < 8; ++j) {
      int row = m0 + wave * 16 + ((lane < 16) ? j : j + 8);
      int col = n0 + t * 16 + (lane & 15);
      float v = acc[t][j] + bias[col];
      if (ACT == 1) v = sigmoidf_(v);
      else if (ACT == 2) v = v * sigmoidf_(v);
      if (OUT16) ((_Float16*)Cv)[(size_t)row * Nn + col] = (_Float16)v;
      else       ((float*)Cv)[(size_t)row * Nn + col] = v;
    }
}

__global__ __launch_bounds__(256) void to_f16(const float* __restrict__ in,
                                              _Float16* __restrict__ out, size_t n) {
  size_t i = (size_t)blockIdx.x * 256 + threadIdx.x;
  if (i < n) out[i] = (_Float16)in[i];
}

__global__ __launch_bounds__(256) void col_means(const float* __restrict__ hs,
                                                 const float* __restrict__ h0,
                                                 const float* __restrict__ h1,
                                                 float* __restrict__ self_s,
                                                 float* __restrict__ others) {
  int t = blockIdx.x * 256 + threadIdx.x;
  int b = t / E_, e = t % E_;
  size_t base = (size_t)b * N_ * E_ + e;
  float a = 0.f, o = 0.f;
  for (int n = 0; n < N_; ++n) {
    size_t idx = base + (size_t)n * E_;
    a += hs[idx];
    o += 0.5f * (h0[idx] + h1[idx]);
  }
  self_s[t] = a * (1.f / N_);
  others[t] = o * (1.f / N_);
}

__global__ __launch_bounds__(256) void make_kv(const float* __restrict__ G,
                                               const float* __restrict__ h0,
                                               const float* __restrict__ h1,
                                               _Float16* __restrict__ kv) {
  size_t idx = (size_t)blockIdx.x * 256 + threadIdx.x;
  size_t b = idx / ((size_t)N_ * E_);
  size_t r = idx - b * (size_t)N_ * E_;
  float g = G[idx];
  kv[b * (size_t)2 * N_ * E_ + r]                   = (_Float16)(h0[idx] * g);
  kv[b * (size_t)2 * N_ * E_ + (size_t)N_ * E_ + r] = (_Float16)(h1[idx] * g);
}

__global__ __launch_bounds__(256) void pair_mean(const float* __restrict__ hs,
                                                 const _Float16* __restrict__ kv,
                                                 float* __restrict__ pair) {
  int t = blockIdx.x * 256 + threadIdx.x;
  int b = t / E_, e = t % E_;
  float a = 0.f;
  for (int n = 0; n < N_; ++n) {
    size_t hi = (size_t)b * N_ * E_ + (size_t)n * E_ + e;
    size_t ki = (size_t)b * 2 * N_ * E_ + (size_t)n * E_ + e;
    a += hs[hi] * (float)kv[ki];
  }
  pair[t] = a * (1.f / N_);
}

__global__ __launch_bounds__(256) void build_qadd(
    const float* __restrict__ self_s, const float* __restrict__ others, const float* __restrict__ pair,
    const float* __restrict__ Wqs, const float* __restrict__ bqs,
    const float* __restrict__ Wqo, const float* __restrict__ bqo,
    const float* __restrict__ Wqp, const float* __restrict__ bqp,
    float* __restrict__ qadd) {
  int t = blockIdx.x * 256 + threadIdx.x;
  int b = t / E_, eo = t % E_;
  float acc = bqs[eo] + bqo[eo] + bqp[eo];
  const float* ws = Wqs + (size_t)eo * E_;
  const float* wo = Wqo + (size_t)eo * E_;
  const float* wp = Wqp + (size_t)eo * E_;
  for (int k = 0; k < E_; ++k)
    acc += self_s[b * E_ + k] * ws[k] + others[b * E_ + k] * wo[k] + pair[b * E_ + k] * wp[k];
  qadd[t] = acc;
}

__global__ __launch_bounds__(256) void build_Q(const float* __restrict__ Qbase,
                                               const float* __restrict__ qadd,
                                               float* __restrict__ Qraw) {
  size_t idx = (size_t)blockIdx.x * 256 + threadIdx.x;
  int b = (int)(idx / (NT_ * E_));
  int te = (int)(idx % (NT_ * E_));
  int e = te % E_;
  Qraw[idx] = Qbase[te] + qadd[b * E_ + e];
}

// Row LayerNorm; OUT16 selects f16 output
template <int OUT16>
__global__ __launch_bounds__(256) void ln_rows(const float* __restrict__ X,
                                               const float* __restrict__ g,
                                               const float* __restrict__ bta,
                                               void* __restrict__ Yv, int cols) {
  int row = blockIdx.x;
  const float* x = X + (size_t)row * cols;
  __shared__ float red[256];
  int tid = threadIdx.x;
  float s = 0.f;
  for (int i = tid; i < cols; i += 256) s += x[i];
  red[tid] = s; __syncthreads();
  for (int st = 128; st > 0; st >>= 1) { if (tid < st) red[tid] += red[tid + st]; __syncthreads(); }
  float mu = red[0] / cols; __syncthreads();
  float v = 0.f;
  for (int i = tid; i < cols; i += 256) { float d = x[i] - mu; v += d * d; }
  red[tid] = v; __syncthreads();
  for (int st = 128; st > 0; st >>= 1) { if (tid < st) red[tid] += red[tid + st]; __syncthreads(); }
  float inv = rsqrtf(red[0] / cols + 1e-5f);
  for (int i = tid; i < cols; i += 256) {
    float y = (x[i] - mu) * inv * g[i] + bta[i];
    if (OUT16) ((_Float16*)Yv)[(size_t)row * cols + i] = (_Float16)y;
    else       ((float*)Yv)[(size_t)row * cols + i] = y;
  }
}

// ---------------------------------------------------------------------------
// Cross-attn scores: S[bh,32,8192] = (Qh 32x64) @ (Kh 8192x64)^T * 0.125
// K is f16, staged via async-to-LDS.  grid (N2/64, B*NH), block 64.
// ---------------------------------------------------------------------------
__global__ __launch_bounds__(64) void attn_scores(const float* __restrict__ Qp,
                                                  const _Float16* __restrict__ Kb,
                                                  float* __restrict__ S) {
  int bh = blockIdx.y;
  int b = bh / NH_, h = bh % NH_;
  int n0 = blockIdx.x * 64;
  __shared__ _Float16 Qs[32][80];   // 160B stride (16B aligned)
  __shared__ _Float16 Ks[64][80];
  int tid = threadIdx.x, lane = tid & 31, wave = tid >> 5;
#pragma unroll
  for (int i = 0; i < 8; ++i) {  // K tile 64x64 halves = 512 chunks
    int c2 = tid + 64 * i;
    int r = c2 >> 3, cc = (c2 & 7) * 8;
    async_copy_b128(&Ks[r][cc], Kb + ((size_t)b * N2_ + n0 + r) * E_ + h * 64 + cc);
  }
  for (int i = tid; i < 32 * 64; i += 64) {
    int r = i >> 6, c = i & 63;
    Qs[r][c] = (_Float16)Qp[((size_t)(b * NT_ + r)) * E_ + h * 64 + c];
  }
  wait_async0();
  __syncthreads();
  int mloc = wave * 16 + (lane & 15);
  int bk0 = (lane < 16) ? 0 : 8;
  int kk0 = (lane < 16) ? 0 : 16;
  v8f acc[4] = {};
#pragma unroll
  for (int ks = 0; ks < 64; ks += 32) {
    v16h a;
#pragma unroll
    for (int i = 0; i < 4; ++i) { a[2 * i] = Qs[mloc][ks + bk0 + 2 * i];      a[2 * i + 1] = Qs[mloc][ks + bk0 + 2 * i + 1]; }
#pragma unroll
    for (int i = 0; i < 4; ++i) { a[8 + 2 * i] = Qs[mloc][ks + bk0 + 16 + 2 * i]; a[9 + 2 * i] = Qs[mloc][ks + bk0 + 16 + 2 * i + 1]; }
#pragma unroll
    for (int t = 0; t < 4; ++t) {
      int n = t * 16 + (lane & 15);
      v16h bf;
#pragma unroll
      for (int j = 0; j < 16; ++j) bf[j] = Ks[n][ks + bk0 + j + (j & 8)];
      acc[t] = __builtin_amdgcn_wmma_f32_16x16x32_f16(false, a, false, bf, (short)0, acc[t], false, false);
    }
  }
#pragma unroll
  for (int t = 0; t < 4; ++t)
#pragma unroll
    for (int j = 0; j < 8; ++j) {
      int row = wave * 16 + ((lane < 16) ? j : j + 8);
      int col = n0 + t * 16 + (lane & 15);
      S[((size_t)bh * NT_ + row) * N2_ + col] = acc[t][j] * 0.125f;
    }
}

__global__ __launch_bounds__(256) void softmax_rows(float* __restrict__ S, int cols) {
  int row = blockIdx.x;
  float* x = S + (size_t)row * cols;
  __shared__ float red[256];
  int tid = threadIdx.x;
  float mx = -1e30f;
  for (int i = tid; i < cols; i += 256) mx = fmaxf(mx, x[i]);
  red[tid] = mx; __syncthreads();
  for (int st = 128; st > 0; st >>= 1) { if (tid < st) red[tid] = fmaxf(red[tid], red[tid + st]); __syncthreads(); }
  mx = red[0]; __syncthreads();
  float s = 0.f;
  for (int i = tid; i < cols; i += 256) { float e = __expf(x[i] - mx); x[i] = e; s += e; }
  red[tid] = s; __syncthreads();
  for (int st = 128; st > 0; st >>= 1) { if (tid < st) red[tid] += red[tid + st]; __syncthreads(); }
  float inv = 1.f / red[0];
  __syncthreads();
  for (int i = tid; i < cols; i += 256) x[i] *= inv;
}

// CTX[b,32,h*64+:64] = S @ Vh.  V f16 staged via async-to-LDS. grid B*NH, block 64.
__global__ __launch_bounds__(64) void attn_av(const float* __restrict__ S,
                                              const _Float16* __restrict__ Vb,
                                              float* __restrict__ CTX) {
  int bh = blockIdx.x;
  int b = bh / NH_, h = bh % NH_;
  __shared__ _Float16 Ss[32][48];
  __shared__ _Float16 Vs[32][80];
  int tid = threadIdx.x, lane = tid & 31, wave = tid >> 5;
  int mloc = wave * 16 + (lane & 15);
  int bk0 = (lane < 16) ? 0 : 8;
  int kk0 = (lane < 16) ? 0 : 16;
  v8f acc[4] = {};
  for (int kc = 0; kc < N2_; kc += 32) {
#pragma unroll
    for (int i = 0; i < 4; ++i) {  // V tile 32x64 halves = 256 chunks
      int c2 = tid + 64 * i;
      int r = c2 >> 3, cc = (c2 & 7) * 8;
      async_copy_b128(&Vs[r][cc], Vb + ((size_t)b * N2_ + kc + r) * E_ + h * 64 + cc);
    }
    for (int i = tid; i < 32 * 32; i += 64) {
      int r = i >> 5, c = i & 31;
      Ss[r][c] = (_Float16)S[((size_t)bh * NT_ + r) * N2_ + kc + c];
    }
    wait_async0();
    __syncthreads();
    v16h a;
#pragma unroll
    for (int i = 0; i < 4; ++i) { a[2 * i] = Ss[mloc][bk0 + 2 * i];      a[2 * i + 1] = Ss[mloc][bk0 + 2 * i + 1]; }
#pragma unroll
    for (int i = 0; i < 4; ++i) { a[8 + 2 * i] = Ss[mloc][bk0 + 16 + 2 * i]; a[9 + 2 * i] = Ss[mloc][bk0 + 16 + 2 * i + 1]; }
#pragma unroll
    for (int t = 0; t < 4; ++t) {
      int n = t * 16 + (lane & 15);
      v16h bf;
#pragma unroll
      for (int j = 0; j < 16; ++j) bf[j] = Vs[bk0 + j + (j & 8)][n];
      acc[t] = __builtin_amdgcn_wmma_f32_16x16x32_f16(false, a, false, bf, (short)0, acc[t], false, false);
    }
    __syncthreads();
  }
#pragma unroll
  for (int t = 0; t < 4; ++t)
#pragma unroll
    for (int j = 0; j < 8; ++j) {
      int row = wave * 16 + ((lane < 16) ? j : j + 8);
      int col = t * 16 + (lane & 15);
      CTX[((size_t)(b * NT_ + row)) * E_ + h * 64 + col] = acc[t][j];
    }
}

// Latent self-attn (32x32), grid B*NH, block 32
__global__ __launch_bounds__(32) void tiny_attn(const float* __restrict__ Q,
                                                const float* __restrict__ K,
                                                const float* __restrict__ V,
                                                float* __restrict__ O) {
  int bh = blockIdx.x;
  int b = bh / NH_, h = bh % NH_;
  __shared__ float Qs[32][64], Ks[32][64], Vs[32][64];
  for (int i = threadIdx.x; i < 32 * 64; i += 32) {
    int r = i >> 6, c = i & 63;
    size_t off = ((size_t)(b * NT_ + r)) * E_ + h * 64 + c;
    Qs[r][c] = Q[off]; Ks[r][c] = K[off]; Vs[r][c] = V[off];
  }
  __syncthreads();
  int q = threadIdx.x;
  float s[32]; float mx = -1e30f;
#pragma unroll
  for (int j = 0; j < 32; ++j) {
    float a = 0.f;
#pragma unroll
    for (int d = 0; d < 64; ++d) a += Qs[q][d] * Ks[j][d];
    s[j] = a * 0.125f; mx = fmaxf(mx, s[j]);
  }
  float sum = 0.f;
#pragma unroll
  for (int j = 0; j < 32; ++j) { s[j] = __expf(s[j] - mx); sum += s[j]; }
  float inv = 1.f / sum;
#pragma unroll
  for (int d = 0; d < 64; ++d) {
    float a = 0.f;
#pragma unroll
    for (int j = 0; j < 32; ++j) a += s[j] * Vs[j][d];
    O[((size_t)(b * NT_ + q)) * E_ + h * 64 + d] = a * inv;
  }
}

__global__ __launch_bounds__(256) void gres_ew(const float* __restrict__ base,
                                               const float* __restrict__ gate,
                                               const float* __restrict__ delta,
                                               const float* __restrict__ scale,
                                               float* __restrict__ out, size_t n) {
  size_t i = (size_t)blockIdx.x * 256 + threadIdx.x;
  if (i < n) out[i] = base[i] + scale[0] * gate[i] * delta[i];
}

// Wide-query attention: f16 queries, f32 K/V (tiny), f16 context out.
__global__ __launch_bounds__(256) void wide_q_attn(const _Float16* __restrict__ Qb,
                                                   const float* __restrict__ Kt,
                                                   const float* __restrict__ Vt,
                                                   _Float16* __restrict__ O) {
  int bh = blockIdx.y;
  int b = bh / NH_, h = bh % NH_;
  __shared__ float Ks[32][64], Vs[32][64];
  for (int i = threadIdx.x; i < 32 * 64; i += 256) {
    int r = i >> 6, c = i & 63;
    size_t off = ((size_t)(b * NT_ + r)) * E_ + h * 64 + c;
    Ks[r][c] = Kt[off]; Vs[r][c] = Vt[off];
  }
  __syncthreads();
  int q = blockIdx.x * 256 + threadIdx.x;
  const _Float16* qp = Qb + ((size_t)b * N_ + q) * E_ + h * 64;
  float qr[64];
#pragma unroll
  for (int d = 0; d < 64; ++d) qr[d] = (float)qp[d];
  float s[32]; float mx = -1e30f;
#pragma unroll
  for (int j = 0; j < 32; ++j) {
    float a = 0.f;
#pragma unroll
    for (int d = 0; d < 64; ++d) a += qr[d] * Ks[j][d];
    s[j] = a * 0.125f; mx = fmaxf(mx, s[j]);
  }
  float sum = 0.f;
#pragma unroll
  for (int j = 0; j < 32; ++j) { s[j] = __expf(s[j] - mx); sum += s[j]; }
  float inv = 1.f / sum;
  _Float16* op = O + ((size_t)b * N_ + q) * E_ + h * 64;
#pragma unroll
  for (int d = 0; d < 64; ++d) {
    float a = 0.f;
#pragma unroll
    for (int j = 0; j < 32; ++j) a += s[j] * Vs[j][d];
    op[d] = (_Float16)(a * inv);
  }
}

// ---------------------------------------------------------------------------
extern "C" void kernel_launch(void* const* d_in, const int* in_sizes, int n_in,
                              void* d_out, int out_size, void* d_ws, size_t ws_size,
                              hipStream_t stream) {
  const float* h_self = (const float*)d_in[0];
  const float* h0     = (const float*)d_in[1];
  const float* h1     = (const float*)d_in[2];
  const float* Qbase  = (const float*)d_in[3];
  const float* Wqs = (const float*)d_in[4];  const float* bqs = (const float*)d_in[5];
  const float* Wqo = (const float*)d_in[6];  const float* bqo = (const float*)d_in[7];
  const float* Wqp = (const float*)d_in[8];  const float* bqp = (const float*)d_in[9];
  const float* qn_g = (const float*)d_in[10]; const float* qn_b = (const float*)d_in[11];
  const float* Wog = (const float*)d_in[12]; const float* bog = (const float*)d_in[13];
  const float* cWq = (const float*)d_in[14]; const float* cbq = (const float*)d_in[15];
  const float* cWk = (const float*)d_in[16]; const float* cbk = (const float*)d_in[17];
  const float* cWv = (const float*)d_in[18]; const float* cbv = (const float*)d_in[19];
  const float* cWo = (const float*)d_in[20]; const float* cbo = (const float*)d_in[21];
  const float* ln_g = (const float*)d_in[22]; const float* ln_b = (const float*)d_in[23];
  const float* lWq = (const float*)d_in[24]; const float* lbq = (const float*)d_in[25];
  const float* lWk = (const float*)d_in[26]; const float* lbk = (const float*)d_in[27];
  const float* lWv = (const float*)d_in[28]; const float* lbv = (const float*)d_in[29];
  const float* lWo = (const float*)d_in[30]; const float* lbo = (const float*)d_in[31];
  const float* fn_g = (const float*)d_in[32]; const float* fn_b = (const float*)d_in[33];
  const float* W1 = (const float*)d_in[34]; const float* b1 = (const float*)d_in[35];
  const float* W2 = (const float*)d_in[36]; const float* b2 = (const float*)d_in[37];
  const float* sc1 = (const float*)d_in[38]; const float* r1Wg = (const float*)d_in[39]; const float* r1bg = (const float*)d_in[40];
  const float* sc2 = (const float*)d_in[41]; const float* r2Wg = (const float*)d_in[42]; const float* r2bg = (const float*)d_in[43];
  const float* on_g = (const float*)d_in[44]; const float* on_b = (const float*)d_in[45];
  const float* oWq = (const float*)d_in[46]; const float* obq = (const float*)d_in[47];
  const float* oWk = (const float*)d_in[48]; const float* obk = (const float*)d_in[49];
  const float* oWv = (const float*)d_in[50]; const float* obv = (const float*)d_in[51];
  const float* oWo = (const float*)d_in[52]; const float* obo = (const float*)d_in[53];
  const float* scO = (const float*)d_in[54]; const float* rOWg = (const float*)d_in[55]; const float* rObg = (const float*)d_in[56];

  const size_t nBNE  = (size_t)B_ * N_ * E_;        // 16.78M
  const size_t nKVE  = (size_t)B_ * 2 * N_ * E_;    // 33.55M
  const size_t nEE   = (size_t)E_ * E_;

  // ---- workspace layout (byte cursor, 256B aligned slots) ----
  char* base = (char*)d_ws;
  size_t cur = 0;
  auto alloc = [&](size_t bytes) { void* p = base + cur; cur += (bytes + 255) & ~(size_t)255; return p; };
  _Float16* kv16  = (_Float16*)alloc(nKVE * 2);   // gated kv, f16
  _Float16* KB16  = (_Float16*)alloc(nKVE * 2);   // cross K proj, f16
  _Float16* VB16  = (_Float16*)alloc(nKVE * 2);   // cross V proj, f16
  float*    bigA  = (float*)   alloc(nBNE * 4);   // G -> S -> gateO (sequential reuse)
  float*    oproj = (float*)   alloc(nBNE * 4);
  _Float16* hs16  = (_Float16*)alloc(nBNE * 2);   // h_self in f16
  _Float16* hn16  = (_Float16*)alloc(nBNE * 2);   // LN(h_self) f16; reused as octx16
  _Float16* Qo16  = (_Float16*)alloc(nBNE * 2);   // out-Q proj, f16
  _Float16* octx16 = hn16;                        // hn16 dead after Qo16 GEMM
  _Float16* Wog16 = (_Float16*)alloc(nEE * 2);
  _Float16* cWk16 = (_Float16*)alloc(nEE * 2);
  _Float16* cWv16 = (_Float16*)alloc(nEE * 2);
  _Float16* oWq16 = (_Float16*)alloc(nEE * 2);
  _Float16* oWo16 = (_Float16*)alloc(nEE * 2);
  _Float16* rOWg16 = (_Float16*)alloc(nEE * 2);
  float* self_s = (float*)alloc(B_ * E_ * 4);
  float* others = (float*)alloc(B_ * E_ * 4);
  float* pair   = (float*)alloc(B_ * E_ * 4);
  float* qadd   = (float*)alloc(B_ * E_ * 4);
  float* Qraw = (float*)alloc(B_ * NT_ * E_ * 4);
  float* Qln  = (float*)alloc(B_ * NT_ * E_ * 4);
  float* Qp   = (float*)alloc(B_ * NT_ * E_ * 4);
  float* ctx  = (float*)alloc(B_ * NT_ * E_ * 4);
  float* tokens = (float*)alloc(B_ * NT_ * E_ * 4);
  float* tin  = (float*)alloc(B_ * NT_ * E_ * 4);
  float* tq = (float*)alloc(B_ * NT_ * E_ * 4);
  float* tk = (float*)alloc(B_ * NT_ * E_ * 4);
  float* tv = (float*)alloc(B_ * NT_ * E_ * 4);
  float* tctx  = (float*)alloc(B_ * NT_ * E_ * 4);
  float* tattn = (float*)alloc(B_ * NT_ * E_ * 4);
  float* gbuf  = (float*)alloc(B_ * NT_ * E_ * 4);
  float* tok2  = (float*)alloc(B_ * NT_ * E_ * 4);
  float* fbuf  = (float*)alloc(B_ * NT_ * E_ * 4);
  float* f1    = (float*)alloc(B_ * NT_ * 2 * E_ * 4);
  float* f2    = (float*)alloc(B_ * NT_ * E_ * 4);
  float* tok3  = (float*)alloc(B_ * NT_ * E_ * 4);
  float* Ko    = (float*)alloc(B_ * NT_ * E_ * 4);
  float* Vo    = (float*)alloc(B_ * NT_ * E_ * 4);

  const int MBN = B_ * N_;      // 32768
  const int MKV = B_ * 2 * N_;  // 65536
  const int MT  = B_ * NT_;     // 256

  // 0) f16 conversions (h_self + big weights)
  to_f16<<<(int)(nBNE / 256), 256, 0, stream>>>(h_self, hs16, nBNE);
  to_f16<<<(int)(nEE / 256), 256, 0, stream>>>(Wog, Wog16, nEE);
  to_f16<<<(int)(nEE / 256), 256, 0, stream>>>(cWk, cWk16, nEE);
  to_f16<<<(int)(nEE / 256), 256, 0, stream>>>(cWv, cWv16, nEE);
  to_f16<<<(int)(nEE / 256), 256, 0, stream>>>(oWq, oWq16, nEE);
  to_f16<<<(int)(nEE / 256), 256, 0, stream>>>(oWo, oWo16, nEE);
  to_f16<<<(int)(nEE / 256), 256, 0, stream>>>(rOWg, rOWg16, nEE);

  // 1) means + gate GEMM + kv + pair
  col_means<<<(B_ * E_) / 256, 256, 0, stream>>>(h_self, h0, h1, self_s, others);
  gemm_f16a<1, 0><<<dim3(E_ / 64, MBN / 128), 256, 0, stream>>>(hs16, Wog16, bog, bigA /*G*/, MBN, E_, E_);
  make_kv<<<(int)(nBNE / 256), 256, 0, stream>>>(bigA, h0, h1, kv16);
  pair_mean<<<(B_ * E_) / 256, 256, 0, stream>>>(h_self, kv16, pair);

  // 2) Q construction + LN
  build_qadd<<<(B_ * E_) / 256, 256, 0, stream>>>(self_s, others, pair, Wqs, bqs, Wqo, bqo, Wqp, bqp, qadd);
  build_Q<<<(B_ * NT_ * E_) / 256, 256, 0, stream>>>(Qbase, qadd, Qraw);
  ln_rows<0><<<MT, 256, 0, stream>>>(Qraw, qn_g, qn_b, Qln, E_);

  // 3) cross attention
  gemm_wmma<0><<<dim3(E_ / 64, MT / 128), 256, 0, stream>>>(Qln, cWq, cbq, Qp, MT, E_, E_);
  gemm_f16a<0, 1><<<dim3(E_ / 64, MKV / 128), 256, 0, stream>>>(kv16, cWk16, cbk, KB16, MKV, E_, E_);
  gemm_f16a<0, 1><<<dim3(E_ / 64, MKV / 128), 256, 0, stream>>>(kv16, cWv16, cbv, VB16, MKV, E_, E_);
  attn_scores<<<dim3(N2_ / 64, B_ * NH_), 64, 0, stream>>>(Qp, KB16, bigA /*S*/);
  softmax_rows<<<B_ * NH_ * NT_, 256, 0, stream>>>(bigA, N2_);
  attn_av<<<B_ * NH_, 64, 0, stream>>>(bigA, VB16, ctx);
  gemm_wmma<0><<<dim3(E_ / 64, MT / 128), 256, 0, stream>>>(ctx, cWo, cbo, tokens, MT, E_, E_);

  // 4) latent MHA + gated residual
  ln_rows<0><<<MT, 256, 0, stream>>>(tokens, ln_g, ln_b, tin, E_);
  gemm_wmma<0><<<dim3(E_ / 64, MT / 128), 256, 0, stream>>>(tin, lWq, lbq, tq, MT, E_, E_);
  gemm_wmma<0><<<dim3(E_ / 64, MT / 128), 256, 0, stream>>>(tin, lWk, lbk, tk, MT, E_, E_);
  gemm_wmma<0><<<dim3(E_ / 64, MT / 128), 256, 0, stream>>>(tin, lWv, lbv, tv, MT, E_, E_);
  tiny_attn<<<B_ * NH_, 32, 0, stream>>>(tq, tk, tv, tctx);
  gemm_wmma<0><<<dim3(E_ / 64, MT / 128), 256, 0, stream>>>(tctx, lWo, lbo, tattn, MT, E_, E_);
  gemm_wmma<1><<<dim3(E_ / 64, MT / 128), 256, 0, stream>>>(tokens, r1Wg, r1bg, gbuf, MT, E_, E_);
  gres_ew<<<(B_ * NT_ * E_) / 256, 256, 0, stream>>>(tokens, gbuf, tattn, sc1, tok2, (size_t)B_ * NT_ * E_);

  // 5) FFN + gated residual
  ln_rows<0><<<MT, 256, 0, stream>>>(tok2, fn_g, fn_b, fbuf, E_);
  gemm_wmma<2><<<dim3(2 * E_ / 64, MT / 128), 256, 0, stream>>>(fbuf, W1, b1, f1, MT, 2 * E_, E_);
  gemm_wmma<0><<<dim3(E_ / 64, MT / 128), 256, 0, stream>>>(f1, W2, b2, f2, MT, E_, 2 * E_);
  gemm_wmma<1><<<dim3(E_ / 64, MT / 128), 256, 0, stream>>>(tok2, r2Wg, r2bg, gbuf, MT, E_, E_);
  gres_ew<<<(B_ * NT_ * E_) / 256, 256, 0, stream>>>(tok2, gbuf, f2, sc2, tok3, (size_t)B_ * NT_ * E_);

  // 6) output MHA + gated residual
  ln_rows<1><<<MBN, 256, 0, stream>>>(h_self, on_g, on_b, hn16, E_);
  gemm_f16a<0, 1><<<dim3(E_ / 64, MBN / 128), 256, 0, stream>>>(hn16, oWq16, obq, Qo16, MBN, E_, E_);
  gemm_wmma<0><<<dim3(E_ / 64, MT / 128), 256, 0, stream>>>(tok3, oWk, obk, Ko, MT, E_, E_);
  gemm_wmma<0><<<dim3(E_ / 64, MT / 128), 256, 0, stream>>>(tok3, oWv, obv, Vo, MT, E_, E_);
  wide_q_attn<<<dim3(N_ / 256, B_ * NH_), 256, 0, stream>>>(Qo16, Ko, Vo, octx16);
  gemm_f16a<0, 0><<<dim3(E_ / 64, MBN / 128), 256, 0, stream>>>(octx16, oWo16, obo, oproj, MBN, E_, E_);
  gemm_f16a<1, 0><<<dim3(E_ / 64, MBN / 128), 256, 0, stream>>>(hs16, rOWg16, rObg, bigA /*gateO*/, MBN, E_, E_);
  gres_ew<<<(int)(nBNE / 256), 256, 0, stream>>>(h_self, bigA, oproj, scO, (float*)d_out, nBNE);
}
